// LSTMMeasurementSelector_53678501265898
// MI455X (gfx1250) — matrix operand order repaired and checked by reading, not compiled, not run on hardware
//
#include <hip/hip_runtime.h>
#include <cstdint>
#include <cstddef>

// ---------------------------------------------------------------------------
// LSTMMeasurementSelector fused forward for MI455X (gfx1250, wave32, WMMA)
//
// B=16384, T=16, H=512 (4H=2048 gates), P=32, IN=17 (padded to 32), NQ=2, NB=3
//
// Each workgroup (256 thr = 8 wave32) owns a 16-row batch tile for the WHOLE
// forward pass; recurrent state (h, c, xs history) lives in LDS. All GEMMs
// use v_wmma_f32_16x16x32_f16. Weights pre-packed to f16 WMMA-B fragments in
// d_ws. Wave w owns hidden channels [64w,64w+64): 4 gates x 4 N-tiles = 16
// v8f accumulators.
//
// v3: - launder only the OFFSET (not the pointer) through asm-volatile, so
//       address-space provenance is kept -> global_load_b128 (not flat_load,
//       which ticked DScnt and fought the LDS path) while still blocking
//       LICM from hoisting+spilling weight fragments across the t-loop.
//     - B-tile addressing refactored so the 16 tiles of one K-step share a
//       single lane-dependent base and differ only by compile-time immediates
//       (fits the signed 24-bit global-load offset) -> kills the per-tile
//       64-bit VGPR address adds seen in v2.
// ---------------------------------------------------------------------------

typedef __attribute__((ext_vector_type(16))) _Float16 v16h;
typedef __attribute__((ext_vector_type(8)))  float    v8f;

#define DEVINL __device__ __forceinline__

union ABFrag {
  v16h v;
  unsigned int u[8];
  uint4 q[2];
};

// ----- problem constants ----------------------------------------------------
constexpr int B_TOT   = 16384;
constexpr int T_STEPS = 16;
constexpr int H_DIM   = 512;
constexpr int TILE    = 16;          // batch rows per workgroup
constexpr int NTHREADS = 256;        // 8 wave32
constexpr size_t RECON_ELEMS = (size_t)B_TOT * T_STEPS * 32;   // 8388608
// ----- workspace layout (_Float16 elements) --------------------------------
constexpr size_t OFF_HH_SEL = 0;                         // 128*16*512 = 1048576
constexpr size_t OFF_IH_SEL = OFF_HH_SEL + 1048576;      // 128*512    = 65536
constexpr size_t OFF_IH0    = OFF_IH_SEL + 65536;
constexpr size_t OFF_HH0    = OFF_IH0    + 65536;
constexpr size_t OFF_HR0    = OFF_HH0    + 65536;        // 2*16*512   = 16384
constexpr size_t OFF_IH1    = OFF_HR0    + 16384;
constexpr size_t OFF_HH1    = OFF_IH1    + 65536;
constexpr size_t OFF_HR1    = OFF_HH1    + 65536;        // 16384
// ----- dynamic LDS layout (bytes) ------------------------------------------
constexpr int SM_HSEL = 0;                  // 16*512 f16 = 16384  (h / h_raw)
constexpr int SM_CA   = SM_HSEL + 16384;    // 16*512 f32 = 32768  (c sel/L0)
constexpr int SM_CB   = SM_CA   + 32768;    // 16*512 f32 = 32768  (c L1)
constexpr int SM_XS   = SM_CB   + 32768;    // 16t*16r*32 f16 = 16384
constexpr int SM_XC   = SM_XS   + 16384;    // 16*32 f16 = 1024    (current x)
constexpr int SM_HP0  = SM_XC   + 1024;     // 16*32 f16 = 1024    (L0 proj h)
constexpr int SM_HP1  = SM_HP0  + 1024;     // 16*32 f16 = 1024    (L1 proj h)
constexpr int SM_PACW = SM_HP1  + 1024;     // 8w*16*32 f32 = 16384 (proj parts)
constexpr int SM_LOG  = SM_PACW + 16384;    // 16*6 f32 (padded)
constexpr int SMEM_BYTES = SM_LOG + 512;    // ~117 KB dynamic LDS

// ---------------------------------------------------------------------------
// Opaque OFFSET identity: blocks LICM from hoisting (and then spilling)
// loop-invariant weight loads across the recurrent t-loop, while keeping the
// base pointer's global address-space provenance intact (so loads lower to
// global_load_*, not flat_load_*).
// ---------------------------------------------------------------------------
DEVINL size_t opq_sz(size_t x) {
  asm volatile("" : "+s"(x));
  return x;
}

// ---------------------------------------------------------------------------
// Weight packing: W[N,K] f32 row-major -> f16 WMMA-B fragments.
// Within a (kt,nt) 32x16 tile: value(lane, e) = W[nt*16 + lane%16]
//                     [kt*32 + (e/8)*16 + (lane/16)*8 + e%8]
// dst[((nt*KT + kt)*32 + lane)*16 + e], zero-padded for k >= K.
// ---------------------------------------------------------------------------
__global__ void pack_wmma_b(const float* __restrict__ W, int N, int K, int KT,
                            _Float16* __restrict__ dst) {
  int idx = blockIdx.x * blockDim.x + threadIdx.x;
  int total = (N / 16) * KT * 32 * 16;
  if (idx >= total) return;
  int e    = idx & 15;
  int lane = (idx >> 4) & 31;
  int kt   = (idx >> 9) % KT;
  int nt   = (idx >> 9) / KT;
  int n = nt * 16 + (lane & 15);
  int k = kt * 32 + ((e & 8) << 1) + ((lane >> 4) << 3) + (e & 7);
  float val = (k < K) ? W[(size_t)n * K + k] : 0.0f;
  dst[idx] = (_Float16)val;
}

// ---------------------------------------------------------------------------
// Fragment loaders (documented CDNA5 WMMA VGPR layouts)
// ---------------------------------------------------------------------------
// A: 16x32 f16 tile from LDS. Lane L: row m = L%16, half = L/16.
DEVINL v16h load_a(const _Float16* hbase, int rowStride, int k0, int lane) {
  const int m = lane & 15;
  const int half = lane >> 4;
  ABFrag a;
  const unsigned int* row = (const unsigned int*)(hbase + m * rowStride);
#pragma unroll
  for (int v = 0; v < 8; ++v) {
    const int K0 = k0 + ((v & 4) << 2) + (half << 3) + ((v & 3) << 1);
    a.u[v] = row[K0 >> 1];
  }
  return a.v;
}

// B: 32x16 f16 tile; `lanebase` already includes lane*16 and the shared
// (wave,kt) tile offset, so each tile differs only by a constant elemOff ->
// global_load_b128 with an immediate offset.
DEVINL v16h load_b_at(const _Float16* lanebase, int elemOff) {
  ABFrag b;
  const uint4* p = (const uint4*)(lanebase + elemOff);
  b.q[0] = p[0];
  b.q[1] = p[1];
  return b.v;
}

DEVINL v8f wmma16(v16h a, v16h b, v8f c) {
  return __builtin_amdgcn_wmma_f32_16x16x32_f16(false, a, false, b,
                                                (short)0, c, false, false);
}

// 4 gates x 4 N-subtiles accumulation for one K-tile (wave owns 64 channels).
// Tile index nt*KT + kt with nt = g*32 + wave*4 + nj decomposes into a
// uniform-per-wave base ((wave*4*KT + kt)*512) and compile-time immediates
// ((g*32 + nj)*KT*512, KT literal at every call site).
DEVINL void gemm_gates(v8f acc[4][4], v16h a, const _Float16* bbase,
                       int wave, int kt, int KT, int lane) {
  const _Float16* lb = bbase + ((size_t)((wave * 4) * KT + kt) << 9) + (lane << 4);
#pragma unroll
  for (int g = 0; g < 4; ++g) {
#pragma unroll
    for (int nj = 0; nj < 4; ++nj) {
      v16h b = load_b_at(lb, ((g * 32 + nj) * KT) << 9);
      acc[g][nj] = wmma16(a, b, acc[g][nj]);
    }
  }
}

DEVINL float sigm(float x) { return 1.0f / (1.0f + __expf(-x)); }

// Apply LSTM nonlinearity for this wave's 64 channels; update c (f32 LDS)
// and write h (f16 LDS). D-frag: lane L -> n_local = L%16, rows m = (L/16)*8+v.
DEVINL void lstm_gate_update(const v8f acc[4][4], const float* bi, const float* bh,
                             float* c_lds, _Float16* h_lds, int wave, int lane) {
  const int n_l = lane & 15;
  const int half = lane >> 4;
#pragma unroll
  for (int nj = 0; nj < 4; ++nj) {
    const int k = wave * 64 + nj * 16 + n_l;
    const float bI = bi[k]        + bh[k];
    const float bF = bi[512 + k]  + bh[512 + k];
    const float bG = bi[1024 + k] + bh[1024 + k];
    const float bO = bi[1536 + k] + bh[1536 + k];
#pragma unroll
    for (int v = 0; v < 8; ++v) {
      const int m = half * 8 + v;
      const float gi = acc[0][nj][v] + bI;
      const float gf = acc[1][nj][v] + bF;
      const float gg = acc[2][nj][v] + bG;
      const float go = acc[3][nj][v] + bO;
      const float cold = c_lds[m * 512 + k];
      const float cnew = sigm(gf) * cold + sigm(gi) * tanhf(gg);
      const float hnew = sigm(go) * tanhf(cnew);
      c_lds[m * 512 + k] = cnew;
      h_lds[m * 512 + k] = (_Float16)hnew;
    }
  }
}

// Projection [16,512] x [512,32]: wave w takes K tiles {2w, 2w+1}, both
// N-tiles; partial sums go to paccW[wave][16][32] (deterministic cross-wave
// reduction afterwards).
DEVINL void proj_partial(const _Float16* hraw, const _Float16* bbase,
                         float* paccW, int wave, int lane) {
  v8f zero = {};
  v8f accp[2];
  accp[0] = zero;
  accp[1] = zero;
#pragma unroll
  for (int kk = 0; kk < 2; ++kk) {
    const int kt = wave * 2 + kk;
    v16h a = load_a(hraw, 512, kt * 32, lane);
    const _Float16* lb = bbase + ((size_t)kt << 9) + (lane << 4);
#pragma unroll
    for (int nt = 0; nt < 2; ++nt) {
      v16h b = load_b_at(lb, (nt * 16) << 9);
      accp[nt] = wmma16(a, b, accp[nt]);
    }
  }
  const int n_l = lane & 15;
  const int half = lane >> 4;
#pragma unroll
  for (int nt = 0; nt < 2; ++nt)
#pragma unroll
    for (int v = 0; v < 8; ++v)
      paccW[wave * 512 + (half * 8 + v) * 32 + nt * 16 + n_l] = accp[nt][v];
}

// ---------------------------------------------------------------------------
// Fused forward kernel: selector (T steps) + reconstructor layers 0/1.
// ---------------------------------------------------------------------------
__global__ __launch_bounds__(NTHREADS)
void fused_lstm_sel(const float* __restrict__ meas,
                    const float* __restrict__ basis_re,
                    const float* __restrict__ basis_im,
                    const float* __restrict__ rho,
                    const float* __restrict__ h0,
                    const float* __restrict__ c0,
                    const float* __restrict__ bases_re,
                    const float* __restrict__ bases_im,
                    const float* __restrict__ bih_sel,
                    const float* __restrict__ bhh_sel,
                    const float* __restrict__ Wproj,
                    const float* __restrict__ bproj,
                    const float* __restrict__ bih0,
                    const float* __restrict__ bhh0,
                    const float* __restrict__ bih1,
                    const float* __restrict__ bhh1,
                    const _Float16* __restrict__ ws,
                    float* __restrict__ out) {
  extern __shared__ char smem[];
  _Float16* hsel   = (_Float16*)(smem + SM_HSEL);   // [16][512] h / h_raw
  float*    cA     = (float*)   (smem + SM_CA);     // [16][512] c (sel, L0)
  float*    cB     = (float*)   (smem + SM_CB);     // [16][512] c (L1)
  _Float16* xs     = (_Float16*)(smem + SM_XS);     // [16t][16r][32]
  _Float16* xcur   = (_Float16*)(smem + SM_XC);     // [16][32] (17 used)
  _Float16* hp0    = (_Float16*)(smem + SM_HP0);    // [16][32]
  _Float16* hp1    = (_Float16*)(smem + SM_HP1);    // [16][32]
  float*    paccW  = (float*)   (smem + SM_PACW);   // [8][16][32]
  float*    logitsL= (float*)   (smem + SM_LOG);    // [16][6]

  const int tid  = threadIdx.x;
  const int lane = tid & 31;
  const int wave = tid >> 5;
  const int base = blockIdx.x * TILE;

  // ---- init: h0/c0 -> LDS, zero x pad -------------------------------------
  for (int e = tid; e < TILE * H_DIM; e += NTHREADS) {
    const int m = e >> 9, k = e & 511;
    hsel[e] = (_Float16)h0[(size_t)(base + m) * H_DIM + k];
    cA[e]   = c0[(size_t)(base + m) * H_DIM + k];
  }
  for (int e = tid; e < TILE * 32; e += NTHREADS) xcur[e] = (_Float16)0.0f;
  __syncthreads();

  // ---- x0 = [measurement, M0_ri(16)]; bases_all[:,0] ----------------------
  if (tid < TILE) {
    const int b = base + tid;
    xcur[tid * 32 + 0] = (_Float16)meas[b];
    float* bout = out + RECON_ELEMS + (size_t)(b * T_STEPS + 0) * 16;
#pragma unroll
    for (int q = 0; q < 2; ++q)
#pragma unroll
      for (int i = 0; i < 2; ++i)
#pragma unroll
        for (int j = 0; j < 2; ++j) {
          const float re = basis_re[(size_t)b * 8 + q * 4 + i * 2 + j];
          const float im = basis_im[(size_t)b * 8 + q * 4 + i * 2 + j];
          const int idx = q * 8 + i * 4 + j * 2;
          xcur[tid * 32 + 1 + idx]     = (_Float16)re;
          xcur[tid * 32 + 1 + idx + 1] = (_Float16)im;
          bout[idx]     = re;
          bout[idx + 1] = im;
        }
  }
  __syncthreads();

  // =========================== selector loop ================================
#pragma unroll 1
  for (int t = 0; t < T_STEPS; ++t) {
    // save xs[t] (f16 [16][32])
    xs[t * 512 + tid]       = xcur[tid];
    xs[t * 512 + 256 + tid] = xcur[256 + tid];
    if (t == T_STEPS - 1) break;
    __syncthreads();

    // fresh (non-hoistable) offsets each timestep: weights stay in L2, never
    // get promoted to registers across iterations and spilled. Pointer base
    // provenance is preserved -> global_load_b128.
    const _Float16* wIH = ws + opq_sz(OFF_IH_SEL);
    const _Float16* wHH = ws + opq_sz(OFF_HH_SEL);
    const float* WprojL = Wproj + opq_sz(0);
    const float* bprojL = bproj + opq_sz(0);
    const float* rhoL   = rho + opq_sz(0);
    const float* basReL = bases_re + opq_sz(0);
    const float* basImL = bases_im + opq_sz(0);
    const float* biL    = bih_sel + opq_sz(0);
    const float* bhL    = bhh_sel + opq_sz(0);

    // gates: g = x @ Wih^T + h @ Whh^T
    v8f zero = {};
    v8f acc[4][4];
#pragma unroll
    for (int g = 0; g < 4; ++g)
#pragma unroll
      for (int nj = 0; nj < 4; ++nj) acc[g][nj] = zero;

    v16h ax = load_a(xcur, 32, 0, lane);
    gemm_gates(acc, ax, wIH, wave, 0, 1, lane);
#pragma unroll 1
    for (int kt = 0; kt < 16; ++kt) {
      if (kt < 15)
        __builtin_prefetch(wHH + ((size_t)((wave * 4) * 16 + kt + 1) << 9), 0, 1);
      v16h ah = load_a(hsel, 512, kt * 32, lane);
      gemm_gates(acc, ah, wHH, wave, kt, 16, lane);
    }
    __syncthreads();  // all waves done reading hsel

    lstm_gate_update(acc, biL, bhL, cA, hsel, wave, lane);
    __syncthreads();  // hsel updated everywhere

    // logits[b,q,n] = h . Wproj[q,n,:] + bproj ; 16 threads per row
    {
      const int r = tid >> 4, p = tid & 15;
#pragma unroll 1
      for (int qn = 0; qn < 6; ++qn) {
        float s = 0.0f;
        for (int kk = 0; kk < 32; ++kk) {
          const int k = p * 32 + kk;
          s += (float)hsel[r * 512 + k] * WprojL[qn * 512 + k];
        }
        s += __shfl_xor(s, 8, 32);
        s += __shfl_xor(s, 4, 32);
        s += __shfl_xor(s, 2, 32);
        s += __shfl_xor(s, 1, 32);
        if (p == 0) logitsL[r * 6 + qn] = s + bprojL[qn];
      }
    }
    __syncthreads();

    // per-row tail: softmax -> M -> kron -> Re Tr(rho K) -> x_{t+1}, bases out
    if (tid < TILE) {
      const int b = base + tid;
      float Mr[2][2][2], Mi[2][2][2];
#pragma unroll
      for (int q = 0; q < 2; ++q) {
        const float l0 = logitsL[tid * 6 + q * 3 + 0];
        const float l1 = logitsL[tid * 6 + q * 3 + 1];
        const float l2 = logitsL[tid * 6 + q * 3 + 2];
        const float mx = fmaxf(l0, fmaxf(l1, l2));
        const float e0 = __expf(l0 - mx), e1 = __expf(l1 - mx), e2 = __expf(l2 - mx);
        const float inv = 1.0f / (e0 + e1 + e2);
        const float p0 = e0 * inv, p1 = e1 * inv, p2 = e2 * inv;
#pragma unroll
        for (int i = 0; i < 2; ++i)
#pragma unroll
          for (int j = 0; j < 2; ++j) {
            const int ij = i * 2 + j;
            Mr[q][i][j] = p0 * basReL[ij] + p1 * basReL[4 + ij] + p2 * basReL[8 + ij];
            Mi[q][i][j] = p0 * basImL[ij] + p1 * basImL[4 + ij] + p2 * basImL[8 + ij];
          }
      }
      // m = Re sum_{a,c} rho[a][c] * K[c][a],  K[i*2+k][j*2+l] = M0[i][j]*M1[k][l]
      float mval = 0.0f;
#pragma unroll
      for (int a = 0; a < 4; ++a)
#pragma unroll
        for (int cI = 0; cI < 4; ++cI) {
          const int i = cI >> 1, k2 = cI & 1, j = a >> 1, l = a & 1;
          const float Kre = Mr[0][i][j] * Mr[1][k2][l] - Mi[0][i][j] * Mi[1][k2][l];
          const float Kim = Mr[0][i][j] * Mi[1][k2][l] + Mi[0][i][j] * Mr[1][k2][l];
          const float rre = rhoL[(size_t)b * 32 + a * 4 + cI];
          const float rim = rhoL[(size_t)b * 32 + 16 + a * 4 + cI];
          mval += rre * Kre - rim * Kim;
        }
      xcur[tid * 32 + 0] = (_Float16)mval;
      float* bout = out + RECON_ELEMS + (size_t)(b * T_STEPS + (t + 1)) * 16;
#pragma unroll
      for (int q = 0; q < 2; ++q)
#pragma unroll
        for (int i = 0; i < 2; ++i)
#pragma unroll
          for (int j = 0; j < 2; ++j) {
            const int idx = q * 8 + i * 4 + j * 2;
            xcur[tid * 32 + 1 + idx]     = (_Float16)Mr[q][i][j];
            xcur[tid * 32 + 1 + idx + 1] = (_Float16)Mi[q][i][j];
            bout[idx]     = Mr[q][i][j];
            bout[idx + 1] = Mi[q][i][j];
          }
    }
    __syncthreads();
  }

  // ======================= reconstructor layers 0/1 =========================
  for (int e = tid; e < TILE * H_DIM; e += NTHREADS) { cA[e] = 0.0f; cB[e] = 0.0f; }
  for (int e = tid; e < TILE * 32; e += NTHREADS) {
    hp0[e] = (_Float16)0.0f;
    hp1[e] = (_Float16)0.0f;
  }
  __syncthreads();

#pragma unroll 1
  for (int t = 0; t < T_STEPS; ++t) {
    // fresh offsets each timestep (see note above)
    const _Float16* wIH0 = ws + opq_sz(OFF_IH0);
    const _Float16* wHH0 = ws + opq_sz(OFF_HH0);
    const _Float16* wHR0 = ws + opq_sz(OFF_HR0);
    const _Float16* wIH1 = ws + opq_sz(OFF_IH1);
    const _Float16* wHH1 = ws + opq_sz(OFF_HH1);
    const _Float16* wHR1 = ws + opq_sz(OFF_HR1);
    const float* bi0 = bih0 + opq_sz(0);
    const float* bh0 = bhh0 + opq_sz(0);
    const float* bi1 = bih1 + opq_sz(0);
    const float* bh1 = bhh1 + opq_sz(0);

    v8f zero = {};
    // ---- layer 0 gates: xs[t] (K=32 incl pad) + hp0 (K=32) ----
    v8f acc[4][4];
#pragma unroll
    for (int g = 0; g < 4; ++g)
#pragma unroll
      for (int nj = 0; nj < 4; ++nj) acc[g][nj] = zero;
    {
      v16h axv = load_a(xs + t * 512, 32, 0, lane);
      gemm_gates(acc, axv, wIH0, wave, 0, 1, lane);
      v16h ahv = load_a(hp0, 32, 0, lane);
      gemm_gates(acc, ahv, wHH0, wave, 0, 1, lane);
    }
    __syncthreads();
    lstm_gate_update(acc, bi0, bh0, cA, hsel, wave, lane);
    __syncthreads();
    // ---- layer 0 projection: y = h_raw @ Whr0^T ----
    proj_partial(hsel, wHR0, paccW, wave, lane);
    __syncthreads();
    for (int e = tid; e < 512; e += NTHREADS) {
      float s = 0.0f;
#pragma unroll
      for (int w = 0; w < 8; ++w) s += paccW[w * 512 + e];
      hp0[e] = (_Float16)s;   // y_t == next-step h for layer0 == layer1 input
    }
    __syncthreads();

    // ---- layer 1 gates: y (hp0) + hp1 ----
#pragma unroll
    for (int g = 0; g < 4; ++g)
#pragma unroll
      for (int nj = 0; nj < 4; ++nj) acc[g][nj] = zero;
    {
      v16h ayv = load_a(hp0, 32, 0, lane);
      gemm_gates(acc, ayv, wIH1, wave, 0, 1, lane);
      v16h ahv = load_a(hp1, 32, 0, lane);
      gemm_gates(acc, ahv, wHH1, wave, 0, 1, lane);
    }
    __syncthreads();
    lstm_gate_update(acc, bi1, bh1, cB, hsel, wave, lane);
    __syncthreads();
    // ---- layer 1 projection -> recon output + hp1 ----
    proj_partial(hsel, wHR1, paccW, wave, lane);
    __syncthreads();
    for (int e = tid; e < 512; e += NTHREADS) {
      float s = 0.0f;
#pragma unroll
      for (int w = 0; w < 8; ++w) s += paccW[w * 512 + e];
      const int m = e >> 5, col = e & 31;
      out[(size_t)((base + m) * T_STEPS + t) * 32 + col] = s;
      hp1[e] = (_Float16)s;
    }
    __syncthreads();
  }
}

// ---------------------------------------------------------------------------
extern "C" void kernel_launch(void* const* d_in, const int* in_sizes, int n_in,
                              void* d_out, int out_size, void* d_ws, size_t ws_size,
                              hipStream_t stream) {
  (void)in_sizes; (void)n_in; (void)out_size; (void)ws_size;
  _Float16* ws = (_Float16*)d_ws;

  const float* Wih_sel = (const float*)d_in[8];
  const float* Whh_sel = (const float*)d_in[9];
  const float* Wih0    = (const float*)d_in[14];
  const float* Whh0    = (const float*)d_in[15];
  const float* Whr0    = (const float*)d_in[18];
  const float* Wih1    = (const float*)d_in[19];
  const float* Whh1    = (const float*)d_in[20];
  const float* Whr1    = (const float*)d_in[23];

  auto packGrid = [](int N, int KT) { return ((N / 16) * KT * 512 + 255) / 256; };
  pack_wmma_b<<<packGrid(2048, 16), 256, 0, stream>>>(Whh_sel, 2048, 512, 16, ws + OFF_HH_SEL);
  pack_wmma_b<<<packGrid(2048, 1),  256, 0, stream>>>(Wih_sel, 2048, 17,  1,  ws + OFF_IH_SEL);
  pack_wmma_b<<<packGrid(2048, 1),  256, 0, stream>>>(Wih0,    2048, 17,  1,  ws + OFF_IH0);
  pack_wmma_b<<<packGrid(2048, 1),  256, 0, stream>>>(Whh0,    2048, 32,  1,  ws + OFF_HH0);
  pack_wmma_b<<<packGrid(32, 16),   256, 0, stream>>>(Whr0,    32,   512, 16, ws + OFF_HR0);
  pack_wmma_b<<<packGrid(2048, 1),  256, 0, stream>>>(Wih1,    2048, 32,  1,  ws + OFF_IH1);
  pack_wmma_b<<<packGrid(2048, 1),  256, 0, stream>>>(Whh1,    2048, 32,  1,  ws + OFF_HH1);
  pack_wmma_b<<<packGrid(32, 16),   256, 0, stream>>>(Whr1,    32,   512, 16, ws + OFF_HR1);

  fused_lstm_sel<<<B_TOT / TILE, NTHREADS, SMEM_BYTES, stream>>>(
      (const float*)d_in[0],   // measurement
      (const float*)d_in[1],   // basis_real
      (const float*)d_in[2],   // basis_imag
      (const float*)d_in[3],   // rho
      (const float*)d_in[4],   // h0
      (const float*)d_in[5],   // c0
      (const float*)d_in[6],   // bases_real
      (const float*)d_in[7],   // bases_imag
      (const float*)d_in[10],  // bih_sel
      (const float*)d_in[11],  // bhh_sel
      (const float*)d_in[12],  // Wproj
      (const float*)d_in[13],  // bproj
      (const float*)d_in[16],  // bih0
      (const float*)d_in[17],  // bhh0
      (const float*)d_in[21],  // bih1
      (const float*)d_in[22],  // bhh1
      ws,
      (float*)d_out);
}
